// Bottleneck_15281493639878
// MI455X (gfx1250) — matrix operand order repaired and checked by reading, not compiled, not run on hardware
//
#include <hip/hip_runtime.h>

typedef __attribute__((ext_vector_type(16))) __bf16 v16bf;
typedef __attribute__((ext_vector_type(8)))  __bf16 v8bf;
typedef __attribute__((ext_vector_type(8)))  float  v8f;
typedef __attribute__((ext_vector_type(4)))  float  v4f;

#define S_SPATIAL 32768   // T*D*H*W = 8*16*16*16
#define NBATCH 2
#define PLN 18            // 16 + halo

static __device__ __forceinline__ v8f wmma_bf16(v16bf a, v16bf b, v8f c) {
  // D = A(16x32 bf16) x B(32x16 bf16) + C(16x16 f32)
  return __builtin_amdgcn_wmma_f32_16x16x32_bf16(false, a, false, b, (short)0, c,
                                                 false, false);
}

static __device__ __forceinline__ v16bf ld2x8(const __bf16* p0, const __bf16* p1) {
  union { v16bf v; v8bf p[2]; } u;
  u.p[0] = *reinterpret_cast<const v8bf*>(p0);
  u.p[1] = *reinterpret_cast<const v8bf*>(p1);
  return u.v;
}

// Async 16B copy global -> LDS (CDNA5, ASYNCcnt-tracked; no VGPR round-trip)
static __device__ __forceinline__ void async_cp16(const __bf16* lds_dst,
                                                  const __bf16* gsrc) {
  unsigned lds_off = (unsigned)(uintptr_t)lds_dst;  // low 32 bits = LDS offset
  unsigned long long ga = (unsigned long long)(uintptr_t)gsrc;
  asm volatile("global_load_async_to_lds_b128 %0, %1, off"
               :: "v"(lds_off), "v"(ga) : "memory");
}
static __device__ __forceinline__ void async_wait0() {
  asm volatile("s_wait_asynccnt 0x0" ::: "memory");
}

// ---------------------------------------------------------------------------
// 1x1-conv GEMM, bf16 WMMA, f32 acc, fused BN partial stats.
//   W[M][K] (f32) x B -> Y ; tile 64(M) x 128(N) per 256-thread block.
//   BLAST: B is channel-last  [b][S][K]  (contiguous K -> one 32B fragment load)
//   else : B is channel-major [b][K][S]  (strided scalar loads, used for x f32)
//   YLAST: Y channel-last [b][S][M] bf16, vectorized 16B epilogue stores
//   else : Y channel-major [b][M][S]
// ---------------------------------------------------------------------------
template <bool BLAST, bool YLAST, typename TB, typename TY>
__global__ __launch_bounds__(256)
void gemm_bn_kernel(const float* __restrict__ Wmat,
                    const TB*   __restrict__ Bmat,
                    TY*         __restrict__ Y,
                    float* __restrict__ gsum, float* __restrict__ gsq,
                    int M, int K)
{
  __shared__ float s_sum[64];
  __shared__ float s_sq[64];

  const int tid  = threadIdx.x;
  const int lane = tid & 31;
  const int wave = tid >> 5;
  const int l15  = lane & 15;
  const int h    = lane >> 4;
  const int b    = blockIdx.z;
  const int mt   = blockIdx.y;                              // 64-row tile index
  const int n    = blockIdx.x * 128 + wave * 16 + l15;      // B/D column

  if (tid < 64) { s_sum[tid] = 0.f; s_sq[tid] = 0.f; }
  __syncthreads();

  v8f acc[4];
#pragma unroll
  for (int mb = 0; mb < 4; ++mb)
#pragma unroll
    for (int r = 0; r < 8; ++r) acc[mb][r] = 0.f;

  for (int k0 = 0; k0 < K; k0 += 32) {
    // B fragment: element e -> k = k0 + 16*h + e (column n fixed per lane)
    v16bf bf;
    if constexpr (BLAST) {
      bf = *reinterpret_cast<const v16bf*>(
          Bmat + ((size_t)b * S_SPATIAL + n) * K + k0 + 16 * h);
    } else {
      const TB* bp = Bmat + (size_t)b * K * S_SPATIAL +
                     (size_t)(k0 + 16 * h) * S_SPATIAL + n;
#pragma unroll
      for (int e = 0; e < 16; ++e) bf[e] = (__bf16)(float)bp[(size_t)e * S_SPATIAL];
    }

#pragma unroll
    for (int mb = 0; mb < 4; ++mb) {
      // A fragment: row m = mt*64 + mb*16 + l15; k = 16h..16h+15
      const float* ap = Wmat + (size_t)(mt * 64 + mb * 16 + l15) * K + k0;
      v16bf af;
#pragma unroll
      for (int e = 0; e < 8; ++e) af[e] = (__bf16)ap[16 * h + e];
#pragma unroll
      for (int e = 0; e < 8; ++e) af[8 + e] = (__bf16)ap[16 * h + 8 + e];
      acc[mb] = wmma_bf16(af, bf, acc[mb]);
    }
  }

  // Epilogue: store + BN partial stats (D layout: col = l15, row = 8h + r)
#pragma unroll
  for (int mb = 0; mb < 4; ++mb) {
    if constexpr (YLAST) {
      v8bf vec;
#pragma unroll
      for (int r = 0; r < 8; ++r) {
        float v = acc[mb][r];
        atomicAdd(&s_sum[mb * 16 + 8 * h + r], v);
        atomicAdd(&s_sq[mb * 16 + 8 * h + r],  v * v);
        vec[r] = (__bf16)v;
      }
      *reinterpret_cast<v8bf*>(
          Y + ((size_t)b * S_SPATIAL + n) * M + mt * 64 + mb * 16 + 8 * h) = vec;
    } else {
#pragma unroll
      for (int r = 0; r < 8; ++r) {
        float v   = acc[mb][r];
        int  mloc = mb * 16 + 8 * h + r;
        atomicAdd(&s_sum[mloc], v);
        atomicAdd(&s_sq[mloc],  v * v);
        Y[(size_t)(b * M + mt * 64 + mloc) * S_SPATIAL + n] = (TY)v;
      }
    }
  }
  __syncthreads();
  if (tid < 64) {
    atomicAdd(&gsum[mt * 64 + tid], s_sum[tid]);
    atomicAdd(&gsq[mt * 64 + tid],  s_sq[tid]);
  }
}

// ---------------------------------------------------------------------------
// 4-D conv 3x3x3x3, 64->64 ch, pad 1. One block per (b,t,d): M=64, N=256(HW),
// K = 64ch x 81 taps. a1/y2 channel-last [b][t][d][h][w][64].
// Outer loop over 32-channel K-halves (cb); per (kt,kd): stage zero-padded
// 18x18x32 plane (interior via async global->LDS b128) + the full 9-tap
// weight slice, then run the 9-tap inner loop barrier-free from LDS.
// ---------------------------------------------------------------------------
__global__ __launch_bounds__(256)
void conv4d3_kernel(const __bf16* __restrict__ a1,   // [2][8][16][256][64] bf16
                    const float*  __restrict__ w2,   // [64][64][3][3][3][3] f32
                    __bf16*       __restrict__ y2,   // [2][8][16][256][64] bf16
                    float* __restrict__ gsum, float* __restrict__ gsq)
{
  __shared__ __align__(16) __bf16 s_plane[PLN * PLN * 32];  // 20736 B
  __shared__ __align__(16) __bf16 s_w[9 * 64 * 32];         // 36864 B
  __shared__ float s_sum[64];
  __shared__ float s_sq[64];

  const int blk  = blockIdx.x;           // b*128 + t*16 + d
  const int d    = blk & 15;
  const int t    = (blk >> 4) & 7;
  const int b    = blk >> 7;
  const int tid  = threadIdx.x;
  const int lane = tid & 31;
  const int wave = tid >> 5;
  const int l15  = lane & 15;
  const int h    = lane >> 4;

  if (tid < 64) { s_sum[tid] = 0.f; s_sq[tid] = 0.f; }

  v8f acc[4][2];
#pragma unroll
  for (int mb = 0; mb < 4; ++mb)
#pragma unroll
    for (int j = 0; j < 2; ++j)
#pragma unroll
      for (int r = 0; r < 8; ++r) acc[mb][j][r] = 0.f;

  for (int cb = 0; cb < 2; ++cb) {
    for (int kt = 0; kt < 3; ++kt) {
      const int ts = t + kt - 1;
      for (int kd = 0; kd < 3; ++kd) {
        const int dz   = d + kd - 1;
        const bool inb = (ts >= 0) & (ts < 8) & (dz >= 0) & (dz < 16);
        __syncthreads();  // prior consumers of s_plane/s_w done

        // ---- Stage plane: zero halo ring (or whole plane if t/d OOB) ----
        for (int p = tid; p < PLN * PLN; p += 256) {
          int yy = p / PLN, xx = p % PLN;
          bool ring = (yy == 0) | (yy == PLN - 1) | (xx == 0) | (xx == PLN - 1);
          if (ring || !inb) {
            v8bf z;
#pragma unroll
            for (int e = 0; e < 8; ++e) z[e] = (__bf16)0.f;
            v8bf* dst = reinterpret_cast<v8bf*>(&s_plane[p * 32]);
            dst[0] = z; dst[1] = z; dst[2] = z; dst[3] = z;
          }
        }
        // ---- Interior: async global->LDS, 16B per lane-op ----
        if (inb) {
          const __bf16* gp =
              a1 + ((size_t)((b * 8 + ts) * 16 + dz) * 256) * 64 + cb * 32;
          for (int q = tid; q < 1024; q += 256) {      // 256 px * 4 chunks
            int pix = q >> 2, ch = q & 3;
            int hy = pix >> 4, wx = pix & 15;
            async_cp16(&s_plane[((hy + 1) * PLN + (wx + 1)) * 32 + ch * 8],
                       gp + (size_t)pix * 64 + ch * 8);
          }
          async_wait0();
        }
        // ---- Stage 9-tap weight slice: s_w[t9][o][ci32] ----
        const int tapbase = (kt * 3 + kd) * 9;
        for (int i = tid; i < 9 * 64 * 32; i += 256) {
          int ci = i & 31, o = (i >> 5) & 63, t9 = i >> 11;
          s_w[i] = (__bf16)w2[(size_t)(o * 64 + cb * 32 + ci) * 81 + tapbase + t9];
        }
        __syncthreads();

        // ---- Inner 9-tap loop: pure LDS + WMMA, no barriers ----
#pragma unroll
        for (int kh = 0; kh < 3; ++kh) {
#pragma unroll
          for (int kw = 0; kw < 3; ++kw) {
            const int t9 = kh * 3 + kw;
            v16bf bfrag[2];
#pragma unroll
            for (int j = 0; j < 2; ++j) {
              const int n  = wave * 32 + j * 16 + l15;   // output pixel
              const int ry = (n >> 4) + kh;              // 0..17 (halo)
              const int rx = (n & 15) + kw;
              bfrag[j] = *reinterpret_cast<const v16bf*>(
                  &s_plane[(ry * PLN + rx) * 32 + 16 * h]);
            }
#pragma unroll
            for (int mb = 0; mb < 4; ++mb) {
              const __bf16* abase = &s_w[(t9 * 64 + mb * 16 + l15) * 32];
              v16bf af = ld2x8(abase + 16 * h, abase + 16 * h + 8);
#pragma unroll
              for (int j = 0; j < 2; ++j)
                acc[mb][j] = wmma_bf16(af, bfrag[j], acc[mb][j]);
            }
          }
        }
      }
    }
  }

  // Epilogue: store y2 (channel-last, 16B vectors) + BN partial stats
#pragma unroll
  for (int mb = 0; mb < 4; ++mb) {
#pragma unroll
    for (int j = 0; j < 2; ++j) {
      const int n = wave * 32 + j * 16 + l15;
      v8bf vec;
#pragma unroll
      for (int r = 0; r < 8; ++r) {
        float v = acc[mb][j][r];
        atomicAdd(&s_sum[mb * 16 + 8 * h + r], v);
        atomicAdd(&s_sq[mb * 16 + 8 * h + r],  v * v);
        vec[r] = (__bf16)v;
      }
      *reinterpret_cast<v8bf*>(
          y2 + ((size_t)((b * 8 + t) * 16 + d) * 256 + n) * 64 + mb * 16 + 8 * h) = vec;
    }
  }
  __syncthreads();
  if (tid < 64) {
    atomicAdd(&gsum[tid], s_sum[tid]);
    atomicAdd(&gsq[tid],  s_sq[tid]);
  }
}

// ---------------------------------------------------------------------------
// Helpers: stats zero, BN finalize, BN apply+ReLU (channel-last), residual.
// ---------------------------------------------------------------------------
__global__ void zero_stats_kernel(float* s) {
  int i = blockIdx.x * 256 + threadIdx.x;
  if (i < 2048) s[i] = 0.f;
}

__global__ void bn_finalize_kernel(const float* __restrict__ sum,
                                   const float* __restrict__ sq,
                                   const float* __restrict__ g,
                                   const float* __restrict__ bb,
                                   float* __restrict__ scale,
                                   float* __restrict__ shift,
                                   int C, float invN)
{
  int c = blockIdx.x * blockDim.x + threadIdx.x;
  if (c < C) {
    float m = sum[c] * invN;
    float v = sq[c] * invN - m * m;
    float s = g[c] * rsqrtf(v + 1e-5f);
    scale[c] = s;
    shift[c] = bb[c] - m * s;
  }
}

// y,a channel-last [b][S][64]: 8 consecutive elems = 8 consecutive channels
__global__ __launch_bounds__(256)
void bn_apply_relu_kernel(const __bf16* __restrict__ y, __bf16* __restrict__ a,
                          const float* __restrict__ scale,
                          const float* __restrict__ shift)
{
  size_t i = ((size_t)blockIdx.x * 256 + threadIdx.x) * 8;
  int c0 = (int)(i & 63);
  v8bf v = *reinterpret_cast<const v8bf*>(y + i);
  v8bf o;
#pragma unroll
  for (int e = 0; e < 8; ++e) {
    float f = (float)v[e] * scale[c0 + e] + shift[c0 + e];
    o[e] = (__bf16)(f > 0.f ? f : 0.f);
  }
  *reinterpret_cast<v8bf*>(a + i) = o;
}

// y3, x channel-major [b][256][S]; streaming: NT load y3, NT store out
__global__ __launch_bounds__(256)
void residual_kernel(const float* __restrict__ y3, const float* __restrict__ x,
                     float* __restrict__ out,
                     const float* __restrict__ scale,
                     const float* __restrict__ shift)
{
  size_t i = ((size_t)blockIdx.x * 256 + threadIdx.x) * 4;
  int c = (int)(i >> 15) & 255;
  float sc = scale[c], sh = shift[c];
  v4f yv = __builtin_nontemporal_load(reinterpret_cast<const v4f*>(y3 + i));
  v4f xv = *reinterpret_cast<const v4f*>(x + i);
  v4f ov;
  ov.x = fmaxf(yv.x * sc + sh + xv.x, 0.f);
  ov.y = fmaxf(yv.y * sc + sh + xv.y, 0.f);
  ov.z = fmaxf(yv.z * sc + sh + xv.z, 0.f);
  ov.w = fmaxf(yv.w * sc + sh + xv.w, 0.f);
  __builtin_nontemporal_store(ov, reinterpret_cast<v4f*>(out + i));
}

// ---------------------------------------------------------------------------
extern "C" void kernel_launch(void* const* d_in, const int* in_sizes, int n_in,
                              void* d_out, int out_size, void* d_ws, size_t ws_size,
                              hipStream_t stream)
{
  const float* x  = (const float*)d_in[0];
  const float* w1 = (const float*)d_in[1];
  const float* g1 = (const float*)d_in[2];
  const float* b1 = (const float*)d_in[3];
  const float* w2 = (const float*)d_in[4];
  const float* g2 = (const float*)d_in[5];
  const float* b2 = (const float*)d_in[6];
  const float* w3 = (const float*)d_in[7];
  const float* g3 = (const float*)d_in[8];
  const float* b3 = (const float*)d_in[9];

  // Workspace: 8 KB stats + y1/a1/y2/a2 (bf16, 8 MB each) + y3 (f32, 64 MB)
  float* stats = (float*)d_ws;
  float* sum1 = stats +    0; float* sq1 = stats +   64;
  float* sc1  = stats +  128; float* sh1 = stats +  192;
  float* sum2 = stats +  256; float* sq2 = stats +  320;
  float* sc2  = stats +  384; float* sh2 = stats +  448;
  float* sum3 = stats +  512; float* sq3 = stats +  768;
  float* sc3  = stats + 1024; float* sh3 = stats + 1280;

  const size_t NC64 = (size_t)NBATCH * 64 * S_SPATIAL;
  __bf16* y1 = (__bf16*)((char*)d_ws + 8192);
  __bf16* a1 = y1 + NC64;
  __bf16* y2 = a1 + NC64;
  __bf16* a2 = y2 + NC64;
  float*  y3 = (float*)(a2 + NC64);
  float*  out = (float*)d_out;

  const float invN = 1.f / (float)(NBATCH * S_SPATIAL);

  zero_stats_kernel<<<8, 256, 0, stream>>>(stats);

  // Stage 1: conv1x1 256->64 (B = x, channel-major f32; Y = y1 channel-last)
  gemm_bn_kernel<false, true, float, __bf16><<<dim3(256, 1, 2), 256, 0, stream>>>(
      w1, x, y1, sum1, sq1, 64, 256);
  bn_finalize_kernel<<<1, 64, 0, stream>>>(sum1, sq1, g1, b1, sc1, sh1, 64, invN);
  bn_apply_relu_kernel<<<2048, 256, 0, stream>>>(y1, a1, sc1, sh1);

  // Stage 2: 4-D conv 3^4 64->64 (channel-last in/out)
  conv4d3_kernel<<<256, 256, 0, stream>>>(a1, w2, y2, sum2, sq2);
  bn_finalize_kernel<<<1, 64, 0, stream>>>(sum2, sq2, g2, b2, sc2, sh2, 64, invN);
  bn_apply_relu_kernel<<<2048, 256, 0, stream>>>(y2, a2, sc2, sh2);

  // Stage 3: conv1x1 64->256 (B = a2 channel-last bf16; Y = y3 channel-major f32)
  gemm_bn_kernel<true, false, __bf16, float><<<dim3(256, 4, 2), 256, 0, stream>>>(
      w3, a2, y3, sum3, sq3, 256, 64);
  bn_finalize_kernel<<<1, 256, 0, stream>>>(sum3, sq3, g3, b3, sc3, sh3, 256, invN);

  residual_kernel<<<16384, 256, 0, stream>>>(y3, x, out, sc3, sh3);
}
